// CausalSelfAttention_7464653161193
// MI455X (gfx1250) — matrix operand order repaired and checked by reading, not compiled, not run on hardware
//
#include <hip/hip_runtime.h>

// ---------------------------------------------------------------------------
// Causal self-attention forward for MI455X (gfx1250), bf16 WMMA pipeline.
//   B=2, S=2048, D=1024, H=16, HD=64
// Stages: fp32->bf16 convert -> QKV GEMM -> flash attention -> proj GEMM.
// ---------------------------------------------------------------------------

typedef __bf16 bf16_t;
typedef __attribute__((ext_vector_type(16))) __bf16 v16bf;
typedef __attribute__((ext_vector_type(8)))  __bf16 v8bf;
typedef __attribute__((ext_vector_type(8)))  float  v8f;

constexpr int kB  = 2;
constexpr int kS  = 2048;
constexpr int kD  = 1024;
constexpr int kH  = 16;
constexpr int kHD = 64;
constexpr float kScale = 0.125f; // 1/sqrt(64)

constexpr int kM  = kB * kS;     // 4096 rows for both GEMMs
constexpr int kNQ = 3 * kD;      // 3072
constexpr int kK  = kD;          // 1024

// workspace layout (bf16 element offsets)
constexpr size_t OFF_XB    = 0;                         // B*S*D
constexpr size_t OFF_WQKV  = OFF_XB    + (size_t)kM * kD;
constexpr size_t OFF_WPROJ = OFF_WQKV  + (size_t)kNQ * kD;
constexpr size_t OFF_Q     = OFF_WPROJ + (size_t)kD * kD;
constexpr size_t OFF_K     = OFF_Q     + (size_t)kM * kD;
constexpr size_t OFF_V     = OFF_K     + (size_t)kM * kD;
constexpr size_t OFF_O     = OFF_V     + (size_t)kM * kD;

// ---------------------------------------------------------------------------
// helpers
// ---------------------------------------------------------------------------
__device__ __forceinline__ bf16_t f2bf(float f) {
  unsigned u = __builtin_bit_cast(unsigned, f);
  unsigned r = (u + 0x7FFFu + ((u >> 16) & 1u)) >> 16;
  unsigned short s = (unsigned short)r;
  return __builtin_bit_cast(bf16_t, s);
}

// A-fragment (16x32 bf16, MxK) from row-major [16][ld] region.
// lane = m (0..15); lanes 16..31 hold the same rows' other K-halves:
//   elems 0..7  -> K = (lane>=16 ? 8 : 0) + j
//   elems 8..15 -> K = 16 + (lane>=16 ? 8 : 0) + (j-8)
__device__ __forceinline__ v16bf load_a_frag(const bf16_t* p, int ld) {
  int lane = threadIdx.x & 31;
  int m  = lane & 15;
  int kh = (lane >> 4) * 8;
  const bf16_t* row = p + (size_t)m * ld;
  v8bf lo = *(const v8bf*)(row + kh);
  v8bf hi = *(const v8bf*)(row + 16 + kh);
  v16bf r;
#pragma unroll
  for (int i = 0; i < 8; ++i) { r[i] = lo[i]; r[i + 8] = hi[i]; }
  return r;
}

// B-fragment (32x16 bf16, KxN) from an [N][K] row-major region
// (i.e. column n's K-run is contiguous).
//   lane&15 = n;  elems 0..7 -> K = (lane>=16 ? 8 : 0) + j, elems 8..15 -> K = 16 + (lane>=16 ? 8 : 0) + (j-8).
__device__ __forceinline__ v16bf load_b_frag(const bf16_t* p, int ld) {
  int lane = threadIdx.x & 31;
  int n  = lane & 15;
  int k0 = (lane >> 4) * 8;
  const bf16_t* row = p + (size_t)n * ld + k0;
  v8bf lo = *(const v8bf*)(row);
  v8bf hi = *(const v8bf*)(row + 16);
  v16bf r;
#pragma unroll
  for (int i = 0; i < 8; ++i) { r[i] = lo[i]; r[i + 8] = hi[i]; }
  return r;
}

__device__ __forceinline__ v8f wmma_bf16(v16bf a, v16bf b, v8f c) {
  return __builtin_amdgcn_wmma_f32_16x16x32_bf16(
      false, a, false, b, (short)0, c, false, false);
}

// ---------------------------------------------------------------------------
// stage 0: fp32 -> bf16 conversion
// ---------------------------------------------------------------------------
__global__ __launch_bounds__(256)
void cvt_f32_bf16_kernel(const float* __restrict__ src,
                         bf16_t* __restrict__ dst, int n) {
  int i = blockIdx.x * blockDim.x + threadIdx.x;
  int stride = gridDim.x * blockDim.x;
  for (; i < n; i += stride) dst[i] = f2bf(src[i]);
}

// ---------------------------------------------------------------------------
// GEMM: C[M,N] = A[M,K] * W[N,K]^T + bias   (A, W bf16; accum fp32)
// block = 256 threads (8 waves), tile 128(M) x 64(N), K-step 32.
// Software-pipelined: next tile's global loads issue before the WMMA chain,
// so global latency hides under compute. Fragments are group-loaded before
// the WMMA chain so DS waits can be partial instead of wait-to-zero.
// ---------------------------------------------------------------------------
template <bool QKV>
__global__ __launch_bounds__(256, 1)
void gemm_bf16_kernel(const bf16_t* __restrict__ A,
                      const bf16_t* __restrict__ W,
                      const float* __restrict__ bias,
                      bf16_t* __restrict__ Qo,
                      bf16_t* __restrict__ Ko,
                      bf16_t* __restrict__ Vo,
                      float* __restrict__ Co,
                      int M, int N, int K) {
  __shared__ bf16_t As[128][40]; // +8 pad, rows stay 16B aligned
  __shared__ bf16_t Bs[64][40];

  int tid  = threadIdx.x;
  int wave = tid >> 5;
  int lane = tid & 31;
  int gm0  = blockIdx.y * 128;
  int gn0  = blockIdx.x * 64;

  // per-thread tile-load coordinates
  int ar = tid >> 1, ac = (tid & 1) * 16;   // A: 128x32, 16 elems/thread
  int br = tid >> 2, bc = (tid & 3) * 8;    // W: 64x32,   8 elems/thread
  const bf16_t* ga = A + (size_t)(gm0 + ar) * K + ac;
  const bf16_t* gw = W + (size_t)(gn0 + br) * K + bc;

  v8f acc[4];
#pragma unroll
  for (int t = 0; t < 4; ++t)
#pragma unroll
    for (int i = 0; i < 8; ++i) acc[t][i] = 0.0f;

  // prologue: load tile k0=0 into registers
  v8bf ra0 = *(const v8bf*)(ga);
  v8bf ra1 = *(const v8bf*)(ga + 8);
  v8bf rb0 = *(const v8bf*)(gw);

  for (int k0 = 0; k0 < K; k0 += 32) {
    // commit staged registers to LDS
    *(v8bf*)&As[ar][ac]     = ra0;
    *(v8bf*)&As[ar][ac + 8] = ra1;
    *(v8bf*)&Bs[br][bc]     = rb0;
    __syncthreads();

    // issue next tile's global loads (latency hides under WMMA chain)
    if (k0 + 32 < K) {
      ra0 = *(const v8bf*)(ga + k0 + 32);
      ra1 = *(const v8bf*)(ga + k0 + 40);
      rb0 = *(const v8bf*)(gw + k0 + 32);
      __builtin_prefetch(ga + k0 + 64, 0, 1); // global_prefetch_b8
      __builtin_prefetch(gw + k0 + 64, 0, 1);
    }

    // group-load fragments, then back-to-back WMMAs
    v16bf af  = load_a_frag(&As[wave * 16][0], 40);
    v16bf bf0 = load_b_frag(&Bs[0][0],  40);
    v16bf bf1 = load_b_frag(&Bs[16][0], 40);
    v16bf bf2 = load_b_frag(&Bs[32][0], 40);
    v16bf bf3 = load_b_frag(&Bs[48][0], 40);
    acc[0] = wmma_bf16(af, bf0, acc[0]);
    acc[1] = wmma_bf16(af, bf1, acc[1]);
    acc[2] = wmma_bf16(af, bf2, acc[2]);
    acc[3] = wmma_bf16(af, bf3, acc[3]);
    __syncthreads();
  }

  // epilogue: D layout -> row = r + 8*(lane>=16), col = lane&15
  int mbase = (lane >> 4) * 8;
  int ncol  = lane & 15;
#pragma unroll
  for (int t = 0; t < 4; ++t) {
    int e = gn0 + t * 16 + ncol;
    float bv = bias[e];
#pragma unroll
    for (int r = 0; r < 8; ++r) {
      int row = gm0 + wave * 16 + mbase + r;
      float v = acc[t][r] + bv;
      if constexpr (QKV) {
        int sel = e / kD;          // 0:q 1:k 2:v
        int c   = e % kD;
        int h   = c / kHD;
        int hd  = c % kHD;
        int b   = row / kS;
        int s   = row % kS;
        bf16_t* dst = (sel == 0) ? Qo : (sel == 1 ? Ko : Vo);
        dst[(((size_t)(b * kH + h) * kS + s) * kHD) + hd] = f2bf(v);
      } else {
        Co[(size_t)row * N + e] = v;
      }
    }
  }
}

// ---------------------------------------------------------------------------
// Flash attention: block = 128 threads (4 waves), one block per
// (b*H+h, 64-query tile); each wave owns 16 query rows. Streams 64-key tiles
// with online softmax. K and V tiles are cooperatively staged in LDS
// (K needs no transpose: [key][hd] is already the B-fragment [N][K] layout;
// V is staged transposed). P is routed through LDS to become an A-fragment.
// Fragment groups are loaded before each WMMA chain for DS-latency overlap.
// ---------------------------------------------------------------------------
__global__ __launch_bounds__(128, 1)
void attn_kernel(const bf16_t* __restrict__ Q,
                 const bf16_t* __restrict__ Km,
                 const bf16_t* __restrict__ Vm,
                 bf16_t* __restrict__ O) {
  __shared__ bf16_t Kt[64][72];       // K tile:   [key][hd]
  __shared__ bf16_t Vt[64][72];       // V^T tile: [hd][key]
  __shared__ bf16_t Pl[4][16][72];    // per-wave P tile [16 rows][64 keys]

  int tid  = threadIdx.x;
  int wave = tid >> 5;
  int lane = tid & 31;
  int bh   = blockIdx.x;          // b*H + h
  int b    = bh / kH;
  int h    = bh % kH;
  int qt   = blockIdx.y;          // 64-row query tile

  const bf16_t* Qb = Q  + (size_t)bh * kS * kHD;
  const bf16_t* Kb = Km + (size_t)bh * kS * kHD;
  const bf16_t* Vb = Vm + (size_t)bh * kS * kHD;

  int qrow0 = qt * 64 + wave * 16;
  int mbase = (lane >> 4) * 8;
  int ncol  = lane & 15;

  // Q fragments for this wave's 16 rows: HD=64 -> two K=32 steps
  v16bf qf[2];
  qf[0] = load_a_frag(Qb + (size_t)qrow0 * kHD,       kHD);
  qf[1] = load_a_frag(Qb + (size_t)qrow0 * kHD + 32,  kHD);

  v8f Oacc[4];
#pragma unroll
  for (int t = 0; t < 4; ++t)
#pragma unroll
    for (int i = 0; i < 8; ++i) Oacc[t][i] = 0.0f;

  float mrow[8], lrow[8];
#pragma unroll
  for (int r = 0; r < 8; ++r) { mrow[r] = -3.0e38f; lrow[r] = 0.0f; }

  for (int j = 0; j <= qt; ++j) {
    __syncthreads(); // previous iteration done with Kt/Vt
    { // stage K tile (straight copy) and V^T tile (transposed)
      int key = tid & 63;
      int hd0 = (tid >> 6) * 32;        // each thread: 32 hd of one key row
      const bf16_t* gk = Kb + (size_t)(j * 64 + key) * kHD + hd0;
      const bf16_t* gv = Vb + (size_t)(j * 64 + key) * kHD + hd0;
#pragma unroll
      for (int c = 0; c < 32; c += 8) {
        *(v8bf*)&Kt[key][hd0 + c] = *(const v8bf*)(gk + c);
        v8bf vv = *(const v8bf*)(gv + c);
#pragma unroll
        for (int i = 0; i < 8; ++i) Vt[hd0 + c + i][key] = vv[i];
      }
      if (j < qt) { // prefetch next key tile
        __builtin_prefetch(gk + 64 * kHD, 0, 1);
        __builtin_prefetch(gv + 64 * kHD, 0, 1);
      }
    }
    __syncthreads();

    // ---- S = (Q K^T) * scale: ks-outer, 4 fragments then 4 WMMAs ------
    v8f Sa[4];
#pragma unroll
    for (int t = 0; t < 4; ++t)
#pragma unroll
      for (int i = 0; i < 8; ++i) Sa[t][i] = 0.0f;
#pragma unroll
    for (int ks = 0; ks < 2; ++ks) {
      v16bf kf0 = load_b_frag(&Kt[0][ks * 32],  72);
      v16bf kf1 = load_b_frag(&Kt[16][ks * 32], 72);
      v16bf kf2 = load_b_frag(&Kt[32][ks * 32], 72);
      v16bf kf3 = load_b_frag(&Kt[48][ks * 32], 72);
      Sa[0] = wmma_bf16(qf[ks], kf0, Sa[0]);
      Sa[1] = wmma_bf16(qf[ks], kf1, Sa[1]);
      Sa[2] = wmma_bf16(qf[ks], kf2, Sa[2]);
      Sa[3] = wmma_bf16(qf[ks], kf3, Sa[3]);
    }

    // scale + causal mask + row max
    float rmax[8];
#pragma unroll
    for (int r = 0; r < 8; ++r) rmax[r] = -3.0e38f;
#pragma unroll
    for (int t = 0; t < 4; ++t) {
      int colg = j * 64 + t * 16 + ncol;
#pragma unroll
      for (int r = 0; r < 8; ++r) {
        int rowg = qrow0 + mbase + r;
        float s = Sa[t][r] * kScale;
        if (colg > rowg) s = -3.0e38f;
        Sa[t][r] = s;
        rmax[r]  = fmaxf(rmax[r], s);
      }
    }

    // row-max across the 16 lanes sharing each row (wave32 halves stay apart)
#pragma unroll
    for (int r = 0; r < 8; ++r)
#pragma unroll
      for (int x = 1; x < 16; x <<= 1)
        rmax[r] = fmaxf(rmax[r], __shfl_xor(rmax[r], x, 32));

    float alpha[8];
#pragma unroll
    for (int r = 0; r < 8; ++r) {
      float mnew = fmaxf(mrow[r], rmax[r]);
      alpha[r] = __expf(mrow[r] - mnew);
      mrow[r]  = mnew;
    }

    // rescale O by alpha (separate loop keeps live ranges short)
#pragma unroll
    for (int t = 0; t < 4; ++t)
#pragma unroll
      for (int r = 0; r < 8; ++r) Oacc[t][r] *= alpha[r];

    // ---- P = exp(S - m) -> bf16 A-tile in LDS, accumulate row sums ----
    float rsum[8];
#pragma unroll
    for (int r = 0; r < 8; ++r) rsum[r] = 0.0f;
#pragma unroll
    for (int t = 0; t < 4; ++t) {
#pragma unroll
      for (int r = 0; r < 8; ++r) {
        float p = __expf(Sa[t][r] - mrow[r]);
        rsum[r] += p;
        Pl[wave][mbase + r][t * 16 + ncol] = f2bf(p);
      }
    }
#pragma unroll
    for (int r = 0; r < 8; ++r) {
#pragma unroll
      for (int x = 1; x < 16; x <<= 1)
        rsum[r] += __shfl_xor(rsum[r], x, 32);
      lrow[r] = lrow[r] * alpha[r] + rsum[r];
    }

    // ---- O += P V: ks-outer, P + 4 V fragments then 4 WMMAs -----------
#pragma unroll
    for (int ks = 0; ks < 2; ++ks) {
      v16bf pf  = load_a_frag(&Pl[wave][0][ks * 32], 72);
      v16bf vf0 = load_b_frag(&Vt[0][ks * 32],  72);
      v16bf vf1 = load_b_frag(&Vt[16][ks * 32], 72);
      v16bf vf2 = load_b_frag(&Vt[32][ks * 32], 72);
      v16bf vf3 = load_b_frag(&Vt[48][ks * 32], 72);
      Oacc[0] = wmma_bf16(pf, vf0, Oacc[0]);
      Oacc[1] = wmma_bf16(pf, vf1, Oacc[1]);
      Oacc[2] = wmma_bf16(pf, vf2, Oacc[2]);
      Oacc[3] = wmma_bf16(pf, vf3, Oacc[3]);
    }
  }

  // finalize: O /= l, store bf16 into [B,S,D] (head-interleaved) layout
#pragma unroll
  for (int t = 0; t < 4; ++t) {
#pragma unroll
    for (int r = 0; r < 8; ++r) {
      float v = Oacc[t][r] / lrow[r];
      int sg = qrow0 + mbase + r;
      O[(size_t)(b * kS + sg) * kD + h * kHD + t * 16 + ncol] = f2bf(v);
    }
  }
}

// ---------------------------------------------------------------------------
// launch
// ---------------------------------------------------------------------------
extern "C" void kernel_launch(void* const* d_in, const int* in_sizes, int n_in,
                              void* d_out, int out_size, void* d_ws, size_t ws_size,
                              hipStream_t stream) {
  const float* x      = (const float*)d_in[0];
  const float* w_qkv  = (const float*)d_in[1];
  const float* b_qkv  = (const float*)d_in[2];
  const float* w_proj = (const float*)d_in[3];
  const float* b_proj = (const float*)d_in[4];
  float* out = (float*)d_out;

  bf16_t* ws     = (bf16_t*)d_ws;
  bf16_t* xb     = ws + OFF_XB;
  bf16_t* wqkvb  = ws + OFF_WQKV;
  bf16_t* wprojb = ws + OFF_WPROJ;
  bf16_t* Qb     = ws + OFF_Q;
  bf16_t* Kb     = ws + OFF_K;
  bf16_t* Vb     = ws + OFF_V;
  bf16_t* Ob     = ws + OFF_O;

  // stage 0: convert inputs to bf16
  cvt_f32_bf16_kernel<<<4096, 256, 0, stream>>>(x,      xb,     kM * kD);
  cvt_f32_bf16_kernel<<<4096, 256, 0, stream>>>(w_qkv,  wqkvb,  kNQ * kD);
  cvt_f32_bf16_kernel<<<2048, 256, 0, stream>>>(w_proj, wprojb, kD * kD);

  // stage 1: QKV projection (M=4096, N=3072, K=1024) -> Q/K/V [B,H,S,HD]
  gemm_bf16_kernel<true><<<dim3(kNQ / 64, kM / 128), 256, 0, stream>>>(
      xb, wqkvb, b_qkv, Qb, Kb, Vb, nullptr, kM, kNQ, kK);

  // stage 2: flash attention -> Ob [B,S,D] bf16
  attn_kernel<<<dim3(kB * kH, kS / 64), 128, 0, stream>>>(Qb, Kb, Vb, Ob);

  // stage 3: output projection (M=4096, N=1024, K=1024) -> fp32 d_out
  gemm_bf16_kernel<false><<<dim3(kD / 64, kM / 128), 256, 0, stream>>>(
      Ob, wprojb, b_proj, nullptr, nullptr, nullptr, out, kM, kD, kK);
}
